// LocalGlobalCrossAttentionModel_75479755260553
// MI455X (gfx1250) — compile-verified, matrix operand order checked by reading, not run on hardware
//
#include <hip/hip_runtime.h>

// ---------------------------------------------------------------------------
// LocalGlobalCrossAttentionModel — MI455X (gfx1250) implementation.
//
// Dead-code insight: softmax over a singleton axis makes attn == 1.0 exactly,
// so the whole shapelet/"local" branch never reaches the output. Live path:
//   x -> conv1+relu+pool -> conv2+relu+pool -> mean -> LN -> v -> out -> MLP
// conv2 (~8 GFLOP) runs as implicit GEMM on WMMA (v_wmma_f32_16x16x32_f16,
// f32 accumulate). One block per (batch, ow-tile) covers all 8 output rows so
// the input tile is read from global exactly once; weights are DMA'd into LDS
// with the Tensor Data Mover (tensor_load_to_lds + s_wait_tensorcnt).
// ---------------------------------------------------------------------------

typedef __attribute__((ext_vector_type(16))) _Float16     v16h;
typedef __attribute__((ext_vector_type(8)))  float        v8f;
typedef __attribute__((ext_vector_type(4)))  unsigned int v4u;
typedef __attribute__((ext_vector_type(8)))  int          v8i;
typedef __attribute__((ext_vector_type(4)))  int          v4i;

#define BATCH 64
#define C1 16      // conv1 out channels
#define C2 32      // conv2 out channels
#define H8 8       // image height
#define W1 2048    // image width
#define W1P 1024   // width after pool1
#define KDIM 240   // conv2 reduction: 16 ic * 3 kh * 5 kw (zero-pad to 256)
#define AF_HALVES (2 * 8 * 32 * 16)   // 8192 f16 weight-fragment elements

// --- prep: swizzle conv2 weights into WMMA A-fragment order (f16), zero g --
// A-fragment layout for v_wmma_f32_16x16x32_f16 (wave32):
//   lane L (0..15): row M = L, halves j<8 -> K = kbase + j, j>=8 -> kbase+16+(j-8)
//   lane L+16     : row M = L, kbase shifted by +8 (hi half)
__global__ void prep_kernel(const float* __restrict__ conv2_w,
                            _Float16* __restrict__ afrag,
                            float* __restrict__ g) {
  int idx = blockIdx.x * 256 + threadIdx.x;   // 8192
  if (idx < BATCH * C2) g[idx] = 0.f;         // zero mean-accumulator every call
  if (idx >= AF_HALVES) return;
  int j    = idx & 15;
  int lane = (idx >> 4) & 31;
  int s    = (idx >> 9) & 7;
  int m    = idx >> 12;
  int row  = lane & 15;
  int hi   = lane >> 4;
  int oc    = m * 16 + row;
  int kbase = s * 32 + hi * 8;
  int kk    = kbase + j + (j >= 8 ? 8 : 0);
  float v = 0.f;
  if (kk < KDIM) {
    int ic = kk / 15, r = kk % 15, kh = r / 5, kw = r % 5;   // one-time cost
    v = conv2_w[((oc * 16 + ic) * 3 + kh) * 5 + kw];
  }
  afrag[idx] = (_Float16)v;
}

// --- conv1 + ReLU + (1,2) maxpool, scalar (only 0.5 GFLOP) -----------------
__global__ void conv1_pool_kernel(const float* __restrict__ x,
                                  const float* __restrict__ w,    // (16,1,3,5)
                                  const float* __restrict__ bias, // (16)
                                  float* __restrict__ h1p) {      // (B,16,8,1024)
  int idx = blockIdx.x * 256 + threadIdx.x;
  if (idx >= BATCH * C1 * H8 * W1P) return;
  int owp = idx & 1023;
  int oh  = (idx >> 10) & 7;
  int oc  = (idx >> 13) & 15;
  int b   = idx >> 17;
  float m = 0.f;                      // post-ReLU values are >= 0
  for (int p = 0; p < 2; ++p) {
    int ow = owp * 2 + p;
    float acc = bias[oc];
    for (int kh = 0; kh < 3; ++kh) {
      int ih = oh - 1 + kh;
      if (ih < 0 || ih >= H8) continue;
      for (int kw = 0; kw < 5; ++kw) {
        int iw = ow - 2 + kw;
        if (iw < 0 || iw >= W1) continue;
        acc = fmaf(w[(oc * 3 + kh) * 5 + kw], x[(b * 8 + ih) * W1 + iw], acc);
      }
    }
    m = fmaxf(m, fmaxf(acc, 0.f));
  }
  h1p[idx] = m;
}

// --- conv2 implicit GEMM on WMMA, one block per (b, ow-tile) ---------------
// 512 threads = 16 waves; wave w handles (oh = w>>1, oc-tile = w&1).
// Phases: A) TDM weights->LDS + raw input tile global->LDS + zero pads
//         B) im2col fragment build (LDS->LDS, loop-counter math only)
//         C) 8x wmma per wave  D) bias/relu/pool/mean epilogue
__global__ void __launch_bounds__(512)
conv2_wmma_kernel(const float* __restrict__ h1p,
                  const _Float16* __restrict__ afrag,  // global, pre-swizzled
                  const float* __restrict__ bias,      // (32)
                  float* __restrict__ g) {             // (B,32)
  int blk = blockIdx.x;
  int owt = blk & 63;
  int b   = blk >> 6;
  int ow0 = owt * 16;

  __shared__ v16h  AF[2 * 8 * 32];        // weight fragments (16 KB) via TDM
  __shared__ float raw[C1 * H8 * 32];     // input tile: 16 ic x 8 rows x 32 cols (16 KB)
  __shared__ v16h  WB[8 * 8 * 32];        // B fragments per oh: 8 oh x 8 steps x 32 lanes (64 KB)
  __shared__ float gacc[C2];

  int tid  = threadIdx.x;                 // 512
  int lane = tid & 31;
  int wav  = tid >> 5;                    // 0..15
  int col  = lane & 15;
  int hi   = lane >> 4;

  // ---- Phase A ----
  if (wav == 0) {
    // Build D# for a 1-D TDM tile: 8192 x 2-byte elements, global -> LDS.
    unsigned long long ga = (unsigned long long)(uintptr_t)afrag;
    unsigned int lds = (unsigned int)(uintptr_t)(void*)&AF[0];  // low 32 bits = LDS offset
    v4u g0 = {1u,                              // count=1, valid, normal mode
              lds,                             // lds_addr
              (unsigned int)(ga & 0xFFFFFFFFu),
              (unsigned int)((ga >> 32) & 0x01FFFFFFu) | (2u << 30)}; // type=2
    v8i g1 = {(int)(1u << 16),                 // data_size=1 (2 bytes), mask=0
              (int)((AF_HALVES & 0xFFFF) << 16),   // tensor_dim0[15:0] @ bit48
              (int)((AF_HALVES >> 16) | (1u << 16)), // tensor_dim0[31:16], tensor_dim1=1
              (int)((AF_HALVES & 0xFFFF) << 16),   // tile_dim0 @ bit112
              (int)1,                          // tile_dim1=1, tile_dim2=0
              (int)AF_HALVES,                  // tensor_dim0_stride[31:0]
              0, 0};
    v4i g2 = {0, 0, 0, 0};
    v4i g3 = {0, 0, 0, 0};
    v8i g4 = {0, 0, 0, 0, 0, 0, 0, 0};     // 6-arg toolchain: extra group, zero-filled
    __builtin_amdgcn_tensor_load_to_lds(g0, g1, g2, g3, g4, 0);
    __builtin_amdgcn_s_wait_tensorcnt(0);
  }
  if (tid < C2) gacc[tid] = 0.f;
  // raw input tile: cols iw0-2 .. iw0+29 (only 0..19 used; 32 for bit-indexing)
  for (int k = 0; k < 8; ++k) {
    int z   = tid + k * 512;                // z = (ic<<8)|(row<<5)|cc
    int cc  = z & 31;
    int row = (z >> 5) & 7;
    int ic  = z >> 8;
    int iw  = ow0 - 2 + cc;
    float v = 0.f;
    if (iw >= 0 && iw < W1P) v = h1p[((b * C1 + ic) * H8 + row) * W1P + iw];
    raw[z] = v;
  }
  // zero the kk in [224,256) pad block of every oh's fragment set (s=7 block;
  // kk 224..239 get overwritten by the fill phase after the barrier)
  _Float16* WBh = (_Float16*)WB;
  for (int k = 0; k < 8; ++k) {
    int z  = tid + k * 512;
    int oh = z >> 9;
    WBh[oh * 4096 + 3584 + (z & 511)] = (_Float16)0.f;
  }
  __syncthreads();

  // ---- Phase B: build B fragments for all 8 oh from LDS raw tile ----
  {
    int q    = tid >> 4;       // 0..31 -> (oh, ic-group)
    int colt = tid & 15;
    int oh   = q >> 2;
    int icg  = q & 3;
    for (int i = 0; i < 4; ++i) {
      int ic = icg * 4 + i;
      for (int kh = 0; kh < 3; ++kh) {
        int ih = oh - 1 + kh;
        bool hok = (ih >= 0) & (ih < H8);
        for (int kw = 0; kw < 5; ++kw) {
          int kk = ic * 15 + kh * 5 + kw;          // loop-counter math only
          float v = hok ? raw[(ic << 8) | (ih << 5) | (colt + kw)] : 0.f;
          int s  = kk >> 5;
          int kp = kk & 31;
          int hh = (kp >> 3) & 1;
          int jj = (kp & 7) | ((kp & 16) >> 1);
          WBh[(((oh * 256) + (s * 32) + (hh << 4) + colt) << 4) | jj] = (_Float16)v;
        }
      }
    }
  }
  __syncthreads();

  // ---- Phase C: MMA (A and B both from LDS, zero index math) ----
  int m  = wav & 1;            // oc tile
  int oh = wav >> 1;
  v8f acc = {};
#pragma unroll
  for (int s = 0; s < 8; ++s) {
    v16h a  = AF[(m * 8 + s) * 32 + lane];
    v16h bf = WB[(oh * 8 + s) * 32 + lane];
    acc = __builtin_amdgcn_wmma_f32_16x16x32_f16(
        /*neg_a=*/false, a, /*neg_b=*/false, bf,
        /*c_mod=*/(short)0, acc, /*reuse_a=*/false, /*reuse_b=*/false);
  }

  // ---- Phase D: bias + ReLU + (1,2) maxpool (shfl_xor 1) + mean reduce ----
#pragma unroll
  for (int v = 0; v < 8; ++v) {
    int oc = m * 16 + v + 8 * hi;
    float val = fmaxf(acc[v] + bias[oc], 0.f);
    float other = __shfl_xor(val, 1, 32);
    if ((col & 1) == 0) {
      float pooled = fmaxf(val, other);
      atomicAdd(&gacc[oc], pooled);               // ds_add_f32
    }
  }
  __syncthreads();
  if (tid < C2)
    atomicAdd(&g[b * C2 + tid], gacc[tid] * (1.f / 4096.f));  // mean over 8*512
}

// --- tail: layernorm(relu(g @ w_glob + b)) -> v -> out proj -> MLP ---------
__global__ void tail_kernel(const float* __restrict__ g,
                            const float* __restrict__ w_glob, const float* __restrict__ b_glob,
                            const float* __restrict__ g_glob, const float* __restrict__ be_glob,
                            const float* __restrict__ in_proj_w, const float* __restrict__ in_proj_b,
                            const float* __restrict__ out_w, const float* __restrict__ out_b,
                            const float* __restrict__ mlp_w1, const float* __restrict__ mlp_b1,
                            const float* __restrict__ mlp_w2, const float* __restrict__ mlp_b2,
                            float* __restrict__ out) {
  int b = blockIdx.x;
  int e = threadIdx.x;  // 64
  __shared__ float gz[32], pre[64], gl[64], vv[64], ao[64], hm[32];
  if (e < 32) gz[e] = g[b * 32 + e];
  __syncthreads();

  float s = b_glob[e];
  for (int f = 0; f < 32; ++f) s = fmaf(gz[f], w_glob[f * 64 + e], s);
  pre[e] = fmaxf(s, 0.f);
  __syncthreads();

  float m = 0.f;
  for (int f = 0; f < 64; ++f) m += pre[f];
  m *= (1.f / 64.f);
  float var = 0.f;
  for (int f = 0; f < 64; ++f) { float d = pre[f] - m; var = fmaf(d, d, var); }
  var *= (1.f / 64.f);
  gl[e] = (pre[e] - m) * rsqrtf(var + 1e-5f) * g_glob[e] + be_glob[e];
  __syncthreads();

  // attn == softmax over singleton == 1  =>  attn_out = v @ out_w.T + out_b
  float sv = in_proj_b[128 + e];
  for (int f = 0; f < 64; ++f) sv = fmaf(gl[f], in_proj_w[(128 + e) * 64 + f], sv);
  vv[e] = sv;
  __syncthreads();

  float sa = out_b[e];
  for (int f = 0; f < 64; ++f) sa = fmaf(vv[f], out_w[e * 64 + f], sa);
  ao[e] = sa;
  __syncthreads();

  if (e < 32) {
    float sh = mlp_b1[e];
    for (int f = 0; f < 64; ++f) sh = fmaf(ao[f], mlp_w1[f * 32 + e], sh);
    hm[e] = fmaxf(sh, 0.f);
  }
  __syncthreads();

  if (e < 10) {
    float so = mlp_b2[e];
    for (int j = 0; j < 32; ++j) so = fmaf(hm[j], mlp_w2[j * 10 + e], so);
    out[b * 10 + e] = so;
  }
}

extern "C" void kernel_launch(void* const* d_in, const int* in_sizes, int n_in,
                              void* d_out, int out_size, void* d_ws, size_t ws_size,
                              hipStream_t stream) {
  (void)in_sizes; (void)n_in; (void)out_size; (void)ws_size;
  const float* x         = (const float*)d_in[0];
  // d_in[1..5]: shapelets / w_loc / b_loc / g_loc / be_loc — dead code (attn==1)
  const float* conv1_w   = (const float*)d_in[6];
  const float* conv1_b   = (const float*)d_in[7];
  const float* conv2_w   = (const float*)d_in[8];
  const float* conv2_b   = (const float*)d_in[9];
  const float* w_glob    = (const float*)d_in[10];
  const float* b_glob    = (const float*)d_in[11];
  const float* g_glob    = (const float*)d_in[12];
  const float* be_glob   = (const float*)d_in[13];
  const float* in_proj_w = (const float*)d_in[14];
  const float* in_proj_b = (const float*)d_in[15];
  const float* out_w     = (const float*)d_in[16];
  const float* out_b     = (const float*)d_in[17];
  const float* mlp_w1    = (const float*)d_in[18];
  const float* mlp_b1    = (const float*)d_in[19];
  const float* mlp_w2    = (const float*)d_in[20];
  const float* mlp_b2    = (const float*)d_in[21];

  char* ws = (char*)d_ws;
  _Float16* afrag = (_Float16*)ws;                 // 8192 halves  (16 KB)
  float*    gbuf  = (float*)(ws + 16384);          // 2048 floats  (8 KB)
  float*    h1p   = (float*)(ws + 16384 + 8192);   // B*16*8*1024 floats (32 MB, L2-resident)

  prep_kernel<<<32, 256, 0, stream>>>(conv2_w, afrag, gbuf);
  conv1_pool_kernel<<<(BATCH * C1 * H8 * W1P) / 256, 256, 0, stream>>>(x, conv1_w, conv1_b, h1p);
  conv2_wmma_kernel<<<BATCH * 64, 512, 0, stream>>>(h1p, afrag, conv2_b, gbuf);
  tail_kernel<<<BATCH, 64, 0, stream>>>(gbuf, w_glob, b_glob, g_glob, be_glob,
                                        in_proj_w, in_proj_b, out_w, out_b,
                                        mlp_w1, mlp_b1, mlp_w2, mlp_b2, (float*)d_out);
}
